// WaggleMoETabTransformer_86835648790609
// MI455X (gfx1250) — matrix operand order repaired and checked by reading, not compiled, not run on hardware
//
#include <hip/hip_runtime.h>
#include <math.h>

// Problem constants (from reference)
#define B_     16384
#define IN_    128
#define D_     512
#define DEPTH_ 3
#define E_     8
#define FF_    2048
#define EH_    256

typedef __attribute__((ext_vector_type(16))) __bf16       v16bf;
typedef __attribute__((ext_vector_type(8)))  float        v8f;
typedef __attribute__((ext_vector_type(4)))  unsigned int u32x4;
typedef int v4i_t __attribute__((vector_size(16)));   // matches builtin signature

union FragBF { v16bf v; u32x4 u[2]; };

#if __has_builtin(__builtin_amdgcn_global_load_async_to_lds_b128)
#define ASYNC_LDS_OK 1
#endif

// 16B global -> LDS copy. On gfx1250 use the async direct-to-LDS path
// (ASYNCcnt, no VGPR round-trip); otherwise plain load+store.
__device__ __forceinline__ void copy16(const unsigned short* g, unsigned short* l) {
#ifdef ASYNC_LDS_OK
  __builtin_amdgcn_global_load_async_to_lds_b128(
      (__attribute__((address_space(1))) v4i_t*)g,
      (__attribute__((address_space(3))) v4i_t*)l, 0, 0);
#else
  *(u32x4*)l = *(const u32x4*)g;
#endif
}

__device__ __forceinline__ void wait_async() {
#ifdef ASYNC_LDS_OK
#if __has_builtin(__builtin_amdgcn_s_wait_asynccnt)
  __builtin_amdgcn_s_wait_asynccnt(0);
#else
  asm volatile("s_wait_asynccnt 0x0" ::: "memory");
#endif
#endif
}

__device__ __forceinline__ unsigned short f2bf(float f) {
  unsigned int u = __float_as_uint(f);
  unsigned int r = (u + 0x7FFFu + ((u >> 16) & 1u)) >> 16;  // RNE
  return (unsigned short)r;
}

__device__ __forceinline__ float gelu_exact(float x) {
  return 0.5f * x * (1.0f + erff(x * 0.70710678118654752f));
}

// ---------------------------------------------------------------------------
// fp32 -> bf16 convert, optional transpose (produces W in [n][k] layout)
// ---------------------------------------------------------------------------
__global__ void convert_f32_bf16(const float* __restrict__ in,
                                 unsigned short* __restrict__ out,
                                 int rows, int cols, int transpose) {
  size_t i = (size_t)blockIdx.x * blockDim.x + threadIdx.x;
  size_t n = (size_t)rows * cols;
  if (i >= n) return;
  if (transpose) {
    int r = (int)(i / cols), c = (int)(i % cols);
    out[(size_t)c * rows + r] = f2bf(in[i]);
  } else {
    out[i] = f2bf(in[i]);
  }
}

__global__ void zero_f32(float* p, int n) {
  int i = blockIdx.x * blockDim.x + threadIdx.x;
  if (i < n) p[i] = 0.0f;
}

// ---------------------------------------------------------------------------
// bf16 WMMA GEMM:  C[M,N] = A[M,K] (bf16) * W^T,  W stored [N][K] bf16.
// Tile: 128x128 per workgroup, BK=64, double-buffered LDS filled via
// global_load_async_to_lds_b128. 8 waves (4x2); each wave owns 32x64
// (2x4 accumulators) -> 16 v_wmma_f32_16x16x32_bf16 per K-step per wave.
// Epilogue: +bias[n], optional exact GELU, optional per-row scale,
//           optional f32 store/accumulate, optional bf16 store.
// ---------------------------------------------------------------------------
template <bool GELU>
__global__ __launch_bounds__(256)
void gemm_bf16(const unsigned short* __restrict__ A,   // M x K
               const unsigned short* __restrict__ W,   // N x K  (B^T layout)
               const float* __restrict__ bias,         // N (may be null)
               float* __restrict__ outF,               // M x N f32 (or null)
               unsigned short* __restrict__ outB,      // M x N bf16 (or null)
               const float* __restrict__ rowscale,     // per-row scale (or null)
               int M, int N, int K, int rsStride, int accumulate) {
  constexpr int BM = 128, BN = 128, BK = 64, PITCH = 72;  // pitch in halves
  __shared__ unsigned short As[2][BM * PITCH];
  __shared__ unsigned short Bs[2][BN * PITCH];

  const int tid  = threadIdx.x;
  const int wave = tid >> 5;
  const int lane = tid & 31;
  const int wm   = wave >> 1;     // 0..3  (M direction, 32 rows each)
  const int wn   = wave & 1;      // 0..1  (N direction, 64 cols each)
  const int m0   = blockIdx.x * BM;
  const int n0   = blockIdx.y * BN;
  const int mrow = lane & 15;     // fragment row (M for A, N for B)
  const int kg   = lane >> 4;     // K-group select per ISA layout

  // Issue async copies of one BK-slab of A and W into LDS stage `st`.
  // 128 rows x 64 halves = 1024 x 16B chunks per tile; 4 chunks/thread each.
  auto stage_in = [&](int k0, int st) {
    #pragma unroll
    for (int i = 0; i < 4; i++) {
      int c = tid + i * 256;
      int r = c >> 3, s = c & 7;
      copy16(A + (size_t)(m0 + r) * K + k0 + s * 8, &As[st][r * PITCH + s * 8]);
    }
    #pragma unroll
    for (int i = 0; i < 4; i++) {
      int c = tid + i * 256;
      int r = c >> 3, s = c & 7;
      copy16(W + (size_t)(n0 + r) * K + k0 + s * 8, &Bs[st][r * PITCH + s * 8]);
    }
  };

  v8f acc[2][4] = {};

  stage_in(0, 0);
  wait_async();
  __syncthreads();

  int st = 0;
  for (int k0 = 0; k0 < K; k0 += BK) {
    if (k0 + BK < K) stage_in(k0 + BK, st ^ 1);  // prefetch next slab (async)

    #pragma unroll
    for (int ks = 0; ks < 2; ks++) {             // two 32-wide K sub-steps
      FragBF a[2], b[4];
      #pragma unroll
      for (int mi = 0; mi < 2; mi++) {
        const unsigned short* p = &As[st][(wm * 32 + mi * 16 + mrow) * PITCH + ks * 32];
        a[mi].u[0] = *(const u32x4*)(p + kg * 8);        // K = kg*8 .. +7
        a[mi].u[1] = *(const u32x4*)(p + 16 + kg * 8);   // K = 16+kg*8 .. +7
      }
      #pragma unroll
      for (int ni = 0; ni < 4; ni++) {
        const unsigned short* p =
            &Bs[st][(wn * 64 + ni * 16 + mrow) * PITCH + ks * 32 + kg * 8];
        b[ni].u[0] = *(const u32x4*)(p);                 // K = kg*8 .. +7
        b[ni].u[1] = *(const u32x4*)(p + 16);            // K = 16+kg*8 .. +7
      }
      #pragma unroll
      for (int mi = 0; mi < 2; mi++)
        #pragma unroll
        for (int ni = 0; ni < 4; ni++)
          acc[mi][ni] = __builtin_amdgcn_wmma_f32_16x16x32_bf16(
              false, a[mi].v, false, b[ni].v, (short)0, acc[mi][ni], false, false);
    }

    wait_async();       // our async copies for st^1 landed in LDS
    __syncthreads();    // everyone done reading st / writing st^1
    st ^= 1;
  }

  // --- epilogue: C layout f32 16x16: elem r -> M = r + 8*kg, N = lane&15 ---
  #pragma unroll
  for (int mi = 0; mi < 2; mi++) {
    #pragma unroll
    for (int ni = 0; ni < 4; ni++) {
      int gn = n0 + wn * 64 + ni * 16 + (lane & 15);
      float bcol = bias ? bias[gn] : 0.0f;
      int gmBase = m0 + wm * 32 + mi * 16 + 8 * kg;
      #pragma unroll
      for (int r = 0; r < 8; r++) {
        int row = gmBase + r;
        float val = acc[mi][ni][r] + bcol;
        if (GELU) val = gelu_exact(val);
        if (rowscale) val *= rowscale[(size_t)row * rsStride];
        size_t off = (size_t)row * N + gn;
        if (outF) {
          if (accumulate) outF[off] += val; else outF[off] = val;
        }
        if (outB) outB[off] = f2bf(val);
      }
    }
  }
}

// ---------------------------------------------------------------------------
// LayerNorm over D=512, wave per row, writes bf16 (feeds next WMMA GEMM)
// ---------------------------------------------------------------------------
__global__ __launch_bounds__(256)
void ln_bf16_kernel(const float* __restrict__ x, const float* __restrict__ w,
                    const float* __restrict__ b, unsigned short* __restrict__ out,
                    int rows) {
  int row  = (int)((blockIdx.x * (size_t)blockDim.x + threadIdx.x) >> 5);
  int lane = threadIdx.x & 31;
  if (row >= rows) return;
  const float* xr = x + (size_t)row * D_;
  float v[16], s = 0.f, s2 = 0.f;
  #pragma unroll
  for (int i = 0; i < 16; i++) {
    float t = xr[lane + 32 * i];
    v[i] = t; s += t; s2 += t * t;
  }
  #pragma unroll
  for (int off = 16; off > 0; off >>= 1) {
    s  += __shfl_xor(s,  off, 32);
    s2 += __shfl_xor(s2, off, 32);
  }
  float mean = s * (1.0f / D_);
  float var  = s2 * (1.0f / D_) - mean * mean;
  float rstd = rsqrtf(var + 1e-5f);
  unsigned short* orow = out + (size_t)row * D_;
  #pragma unroll
  for (int i = 0; i < 16; i++) {
    int idx = lane + 32 * i;
    orow[idx] = f2bf((v[i] - mean) * rstd * w[idx] + b[idx]);
  }
}

// ---------------------------------------------------------------------------
// Router: per-row logits (D=512 x E=8), softmax, eps-mix, top-2 gate, load sums
// ---------------------------------------------------------------------------
__global__ __launch_bounds__(256)
void router_kernel(const float* __restrict__ z, const float* __restrict__ rw,
                   const float* __restrict__ rb, float* __restrict__ gate,
                   float* __restrict__ loadSum, int rows) {
  int row  = (int)((blockIdx.x * (size_t)blockDim.x + threadIdx.x) >> 5);
  int lane = threadIdx.x & 31;
  if (row >= rows) return;
  float p[E_] = {};
  const float* zr = z + (size_t)row * D_;
  for (int k = lane; k < D_; k += 32) {
    float zv = zr[k];
    #pragma unroll
    for (int e = 0; e < E_; e++) p[e] += zv * rw[k * E_ + e];
  }
  #pragma unroll
  for (int off = 16; off > 0; off >>= 1)
    #pragma unroll
    for (int e = 0; e < E_; e++) p[e] += __shfl_xor(p[e], off, 32);

  if (lane == 0) {
    float q[E_], mx = -1e30f;
    #pragma unroll
    for (int e = 0; e < E_; e++) { q[e] = p[e] + rb[e]; mx = fmaxf(mx, q[e]); }
    float se = 0.f;
    #pragma unroll
    for (int e = 0; e < E_; e++) { q[e] = expf(q[e] - mx); se += q[e]; }
    float inv = 1.0f / se;
    #pragma unroll
    for (int e = 0; e < E_; e++) q[e] = 0.9f * q[e] * inv + 0.0125f; // (1-eps)p+eps/E
    int i0 = 0;
    #pragma unroll
    for (int e = 1; e < E_; e++) if (q[e] > q[i0]) i0 = e;
    int i1 = (i0 == 0) ? 1 : 0;
    #pragma unroll
    for (int e = 0; e < E_; e++) if (e != i0 && q[e] > q[i1]) i1 = e;
    float* gr = gate + (size_t)row * E_;
    #pragma unroll
    for (int e = 0; e < E_; e++)
      gr[e] = (e == i0) ? q[i0] : ((e == i1) ? q[i1] : 0.0f);
    #pragma unroll
    for (int e = 0; e < E_; e++) atomicAdd(&loadSum[e], q[e]);
  }
}

__global__ void aux_kernel(const float* __restrict__ loadSum, float* __restrict__ out) {
  if (threadIdx.x == 0 && blockIdx.x == 0) {
    float aux = 0.f;
    #pragma unroll
    for (int e = 0; e < E_; e++) {
      float ld = loadSum[e] * (1.0f / B_);
      aux += ld * logf(ld * (float)E_ + 1e-9f);
    }
    out[0] = aux / logf((float)E_ + 1e-9f);
  }
}

// ---------------------------------------------------------------------------
// Head: logit[b] = LN(z[b]) . head_w + head_b   (wave per row)
// ---------------------------------------------------------------------------
__global__ __launch_bounds__(256)
void head_kernel(const float* __restrict__ z, const float* __restrict__ lnw,
                 const float* __restrict__ lnb, const float* __restrict__ hw,
                 const float* __restrict__ hb, float* __restrict__ out, int rows) {
  int row  = (int)((blockIdx.x * (size_t)blockDim.x + threadIdx.x) >> 5);
  int lane = threadIdx.x & 31;
  if (row >= rows) return;
  const float* zr = z + (size_t)row * D_;
  float v[16], s = 0.f, s2 = 0.f;
  #pragma unroll
  for (int i = 0; i < 16; i++) {
    float t = zr[lane + 32 * i];
    v[i] = t; s += t; s2 += t * t;
  }
  #pragma unroll
  for (int off = 16; off > 0; off >>= 1) {
    s  += __shfl_xor(s,  off, 32);
    s2 += __shfl_xor(s2, off, 32);
  }
  float mean = s * (1.0f / D_);
  float var  = s2 * (1.0f / D_) - mean * mean;
  float rstd = rsqrtf(var + 1e-5f);
  float dot = 0.f;
  #pragma unroll
  for (int i = 0; i < 16; i++) {
    int idx = lane + 32 * i;
    dot += ((v[i] - mean) * rstd * lnw[idx] + lnb[idx]) * hw[idx];
  }
  #pragma unroll
  for (int off = 16; off > 0; off >>= 1) dot += __shfl_xor(dot, off, 32);
  if (lane == 0) out[row] = dot + hb[0];
}

// ---------------------------------------------------------------------------
// Host-side orchestration
// ---------------------------------------------------------------------------
static void launch_gemm(bool gelu, const unsigned short* A, const unsigned short* W,
                        const float* bias, float* outF, unsigned short* outB,
                        const float* rowscale, int M, int N, int K, int rsStride,
                        int accumulate, hipStream_t s) {
  dim3 grid(M / 128, N / 128), blk(256);
  if (gelu)
    gemm_bf16<true><<<grid, blk, 0, s>>>(A, W, bias, outF, outB, rowscale, M, N, K, rsStride, accumulate);
  else
    gemm_bf16<false><<<grid, blk, 0, s>>>(A, W, bias, outF, outB, rowscale, M, N, K, rsStride, accumulate);
}

static void launch_convert(const float* in, unsigned short* out, int rows, int cols,
                           int transpose, hipStream_t s) {
  size_t n = (size_t)rows * cols;
  int blocks = (int)((n + 255) / 256);
  convert_f32_bf16<<<blocks, 256, 0, s>>>(in, out, rows, cols, transpose);
}

extern "C" void kernel_launch(void* const* d_in, const int* in_sizes, int n_in,
                              void* d_out, int out_size, void* d_ws, size_t ws_size,
                              hipStream_t stream) {
  const float* x         = (const float*)d_in[0];
  const float* embed_w   = (const float*)d_in[1];
  const float* embed_b   = (const float*)d_in[2];
  const float* ln1_w     = (const float*)d_in[3];
  const float* ln1_b     = (const float*)d_in[4];
  const float* inproj_w  = (const float*)d_in[5];
  const float* inproj_b  = (const float*)d_in[6];
  const float* outproj_w = (const float*)d_in[7];
  const float* outproj_b = (const float*)d_in[8];
  const float* ln2_w     = (const float*)d_in[9];
  const float* ln2_b     = (const float*)d_in[10];
  const float* ffn_w1    = (const float*)d_in[11];
  const float* ffn_b1    = (const float*)d_in[12];
  const float* ffn_w2    = (const float*)d_in[13];
  const float* ffn_b2    = (const float*)d_in[14];
  const float* router_w  = (const float*)d_in[15];
  const float* router_b  = (const float*)d_in[16];
  const float* exp_w1    = (const float*)d_in[17];
  const float* exp_b1    = (const float*)d_in[18];
  const float* exp_w2    = (const float*)d_in[19];
  const float* exp_b2    = (const float*)d_in[20];
  const float* head_ln_w = (const float*)d_in[21];
  const float* head_ln_b = (const float*)d_in[22];
  const float* head_w    = (const float*)d_in[23];
  const float* head_b    = (const float*)d_in[24];
  float* out = (float*)d_out;

  char* ws = (char*)d_ws;
  size_t cur = 0;
  auto alloc = [&](size_t bytes) -> void* {
    void* p = ws + cur;
    cur = (cur + bytes + 255) & ~(size_t)255;
    return p;
  };
  float*          h       = (float*)alloc((size_t)B_ * D_ * 4);
  unsigned short* g_bf    = (unsigned short*)alloc((size_t)B_ * D_ * 2);
  unsigned short* v_bf    = (unsigned short*)alloc((size_t)B_ * D_ * 2);
  unsigned short* mid_bf  = (unsigned short*)alloc((size_t)B_ * FF_ * 2);
  unsigned short* z_bf    = (unsigned short*)alloc((size_t)B_ * D_ * 2);
  unsigned short* hmid_bf = (unsigned short*)alloc((size_t)B_ * EH_ * 2);
  unsigned short* x_bf    = (unsigned short*)alloc((size_t)B_ * IN_ * 2);
  float*          gate    = (float*)alloc((size_t)B_ * E_ * 4);
  float*          loadSum = (float*)alloc(E_ * 4);
  unsigned short* wEmbT   = (unsigned short*)alloc((size_t)D_ * IN_ * 2);
  unsigned short* wV      = (unsigned short*)alloc((size_t)DEPTH_ * D_ * D_ * 2);
  unsigned short* wO      = (unsigned short*)alloc((size_t)DEPTH_ * D_ * D_ * 2);
  unsigned short* wF1     = (unsigned short*)alloc((size_t)DEPTH_ * FF_ * D_ * 2);
  unsigned short* wF2     = (unsigned short*)alloc((size_t)DEPTH_ * D_ * FF_ * 2);
  unsigned short* wE1     = (unsigned short*)alloc((size_t)E_ * EH_ * D_ * 2);
  unsigned short* wE2     = (unsigned short*)alloc((size_t)E_ * D_ * EH_ * 2);
  (void)ws_size; (void)n_in; (void)in_sizes; (void)out_size;

  // --- weight/activation conversion to bf16, [n][k] layout for W operands ---
  launch_convert(x, x_bf, B_, IN_, 0, stream);
  launch_convert(embed_w, wEmbT, IN_, D_, 1, stream);             // (IN,D)->(D,IN)
  for (int l = 0; l < DEPTH_; l++) {
    launch_convert(inproj_w + ((size_t)l * 3 * D_ + 2 * D_) * D_,
                   wV + (size_t)l * D_ * D_, D_, D_, 0, stream);  // W.T -> as-is
    launch_convert(outproj_w + (size_t)l * D_ * D_,
                   wO + (size_t)l * D_ * D_, D_, D_, 0, stream);  // W.T -> as-is
    launch_convert(ffn_w1 + (size_t)l * D_ * FF_,
                   wF1 + (size_t)l * FF_ * D_, D_, FF_, 1, stream);
    launch_convert(ffn_w2 + (size_t)l * FF_ * D_,
                   wF2 + (size_t)l * D_ * FF_, FF_, D_, 1, stream);
  }
  for (int e = 0; e < E_; e++) {
    launch_convert(exp_w1 + (size_t)e * D_ * EH_,
                   wE1 + (size_t)e * EH_ * D_, D_, EH_, 1, stream);
    launch_convert(exp_w2 + (size_t)e * EH_ * D_,
                   wE2 + (size_t)e * D_ * EH_, EH_, D_, 1, stream);
  }

  const int rowBlocks = (B_ * 32) / 256;  // wave-per-row kernels

  // --- embed: h = x @ embed_w + embed_b ---
  launch_gemm(false, x_bf, wEmbT, embed_b, h, nullptr, nullptr,
              B_, D_, IN_, 0, 0, stream);

  // --- transformer layers ---
  for (int l = 0; l < DEPTH_; l++) {
    ln_bf16_kernel<<<rowBlocks, 256, 0, stream>>>(h, ln1_w + (size_t)l * D_,
                                                  ln1_b + (size_t)l * D_, g_bf, B_);
    launch_gemm(false, g_bf, wV + (size_t)l * D_ * D_,
                inproj_b + (size_t)l * 3 * D_ + 2 * D_,
                nullptr, v_bf, nullptr, B_, D_, D_, 0, 0, stream);
    launch_gemm(false, v_bf, wO + (size_t)l * D_ * D_,
                outproj_b + (size_t)l * D_,
                h, nullptr, nullptr, B_, D_, D_, 0, 1, stream);
    ln_bf16_kernel<<<rowBlocks, 256, 0, stream>>>(h, ln2_w + (size_t)l * D_,
                                                  ln2_b + (size_t)l * D_, g_bf, B_);
    launch_gemm(true, g_bf, wF1 + (size_t)l * FF_ * D_,
                ffn_b1 + (size_t)l * FF_,
                nullptr, mid_bf, nullptr, B_, FF_, D_, 0, 0, stream);
    launch_gemm(false, mid_bf, wF2 + (size_t)l * D_ * FF_,
                ffn_b2 + (size_t)l * D_,
                h, nullptr, nullptr, B_, D_, FF_, 0, 1, stream);
  }

  // --- router on z = h (pre-MoE) ---
  launch_convert(h, z_bf, B_, D_, 0, stream);
  zero_f32<<<1, 32, 0, stream>>>(loadSum, E_);
  router_kernel<<<rowBlocks, 256, 0, stream>>>(h, router_w, router_b,
                                               gate, loadSum, B_);

  // --- experts: h += gate[:,e] * (gelu(z@W1e+b1e) @ W2e + b2e) ---
  for (int e = 0; e < E_; e++) {
    launch_gemm(true, z_bf, wE1 + (size_t)e * EH_ * D_,
                exp_b1 + (size_t)e * EH_,
                nullptr, hmid_bf, nullptr, B_, EH_, D_, 0, 0, stream);
    launch_gemm(false, hmid_bf, wE2 + (size_t)e * D_ * EH_,
                exp_b2 + (size_t)e * D_,
                h, nullptr, gate + e, B_, D_, EH_, E_, 1, stream);
  }

  // --- outputs: logit[0..B-1], aux at [B] ---
  head_kernel<<<rowBlocks, 256, 0, stream>>>(h, head_ln_w, head_ln_b,
                                             head_w, head_b, out, B_);
  aux_kernel<<<1, 32, 0, stream>>>(loadSum, out + B_);
}
